// GNNAnomalyDetector_21603685499209
// MI455X (gfx1250) — matrix operand order repaired and checked by reading, compiled, not run on hardware
//
#include <hip/hip_runtime.h>
#include <hip/hip_bf16.h>

// ---------------------------------------------------------------------------
// GNN anomaly detector: 3x GCN (residual) + 4-head GAT (mean) + MLP heads.
// MI455X / gfx1250: wave32; dense GEMMs via v_wmma_f32_16x16x32_bf16 with
// bf16-staged operands (pure b128 loads + WMMA in the K-loop); edge phases
// via float atomics (global_atomic_add_f32 / global_atomic_max_num_f32).
// ---------------------------------------------------------------------------

typedef __attribute__((ext_vector_type(16))) __bf16 v16bf;
typedef __attribute__((ext_vector_type(8)))  float  v8f;

union BFrag { v16bf v; unsigned int u[8]; uint4 q[2]; };

__device__ __forceinline__ unsigned short f2bf(float f) {
  unsigned int u = __float_as_uint(f);
  u += 0x7FFFu + ((u >> 16) & 1u);          // round-to-nearest-even
  return (unsigned short)(u >> 16);
}
__device__ __forceinline__ float lrelu(float x) { return x > 0.0f ? x : 0.2f * x; }

// ---------------------------------------------------------------------------
// GEMM: out[rows,kout] = A[rows,KIN] @ W[KIN,kout]
//   A:  bf16 row-major (staged by producer kernels)
//   WT: bf16 [kout,KIN] (weights pre-transposed once)
// One wave per 16x16 output tile. KIN templated -> fully unrolled K-loop:
// per k-step exactly 4x global_load_b128 + 1x v_wmma_f32_16x16x32_bf16.
// ---------------------------------------------------------------------------
template <int KIN>
__global__ __launch_bounds__(256) void gemm_wmma_bf16(
    const unsigned short* __restrict__ A, const unsigned short* __restrict__ WT,
    const float* __restrict__ bias, float* __restrict__ outF,
    unsigned short* __restrict__ outBF, int rows, int kout, int relu)
{
  const int wave = (int)((blockIdx.x * blockDim.x + threadIdx.x) >> 5);
  const int lane = (int)(threadIdx.x & 31);
  const int colTiles = kout >> 4;
  const int nTiles = (rows >> 4) * colTiles;      // rows is a multiple of 16
  if (wave >= nTiles) return;                     // wave-uniform: EXEC stays all-1 for WMMA
  const int rt = wave / colTiles;
  const int ct = wave - rt * colTiles;
  const int l15 = lane & 15;
  const int hi  = lane >> 4;                      // ISA half-wave selector
  const int m = rt * 16 + l15;                    // A row this lane owns
  const int n = ct * 16 + l15;                    // B/C column this lane owns
  const unsigned short* arow = A  + (size_t)m * KIN;
  const unsigned short* brow = WT + (size_t)n * KIN;

  v8f acc = {};
#pragma unroll
  for (int k0 = 0; k0 < KIN; k0 += 32) {
    BFrag a, b;
    // A 16x32 bf16 frag: VGPR v<4 -> K = hi*8 + 2v; v>=4 -> K = 16 + hi*8 + 2(v-4)
    a.q[0] = *(const uint4*)(arow + k0 + hi * 8);
    a.q[1] = *(const uint4*)(arow + k0 + 16 + hi * 8);
    // B 32x16 bf16 frag: VGPR v<4 -> K = hi*8 + 2v; v>=4 -> K = 16 + hi*8 + 2(v-4)
    b.q[0] = *(const uint4*)(brow + k0 + hi * 8);
    b.q[1] = *(const uint4*)(brow + k0 + 16 + hi * 8);
    acc = __builtin_amdgcn_wmma_f32_16x16x32_bf16(
        /*neg_a=*/false, a.v, /*neg_b=*/false, b.v,
        /*c_mod=*/(short)0, acc, /*reuse_a=*/false, /*reuse_b=*/false);
  }

  const float bc = bias ? bias[n] : 0.0f;
#pragma unroll
  for (int r = 0; r < 8; ++r) {                   // C/D: VGPR r -> row hi*8+r
    const int row = rt * 16 + hi * 8 + r;
    float v = acc[r] + bc;
    if (relu) v = fmaxf(v, 0.0f);
    const size_t idx = (size_t)row * kout + n;
    if (outF)  outF[idx]  = v;
    if (outBF) outBF[idx] = f2bf(v);
  }
}

// ---------------------------------------------------------------------------
// Setup kernels
// ---------------------------------------------------------------------------
__global__ void degree_kernel(const int* __restrict__ dst, float* __restrict__ deg, int nE) {
  int i = blockIdx.x * blockDim.x + threadIdx.x;
  if (i < nE) atomicAdd(&deg[dst[i]], 1.0f);
}

__global__ void dinv_kernel(float* __restrict__ deg, int n) {
  int i = blockIdx.x * blockDim.x + threadIdx.x;
  if (i < n) deg[i] = rsqrtf(deg[i] + 1.0f);      // +1 = self loop; deg>0 always
}

// f32 row-major -> bf16 row-major (activation staging for the first GEMM)
__global__ void cvt_bf16(const float* __restrict__ A, unsigned short* __restrict__ B, int n) {
  int i = blockIdx.x * blockDim.x + threadIdx.x;
  if (i < n) B[i] = f2bf(A[i]);
}

// W[kin,kout] f32 -> WT[kout,kin] bf16
__global__ void cvt_w_t(const float* __restrict__ W, unsigned short* __restrict__ WT,
                        int kin, int kout) {
  int i = blockIdx.x * blockDim.x + threadIdx.x;
  if (i >= kin * kout) return;
  int r = i / kout, c = i - r * kout;
  WT[(size_t)c * kin + r] = f2bf(W[i]);
}

// ---------------------------------------------------------------------------
// GCN edge aggregation: one wave per edge, float2 per lane (64 features).
// agg[dst] += t[src] * dinv[src]*dinv[dst]
// ---------------------------------------------------------------------------
__global__ __launch_bounds__(256) void gcn_edge_agg(
    const int* __restrict__ src, const int* __restrict__ dst,
    const float* __restrict__ t, const float* __restrict__ dinv,
    float* __restrict__ agg, int nEdges)
{
  const int wave = (int)((blockIdx.x * blockDim.x + threadIdx.x) >> 5);
  const int lane = (int)(threadIdx.x & 31);
  if (wave >= nEdges) return;
  const int s = src[wave], d = dst[wave];
  const float norm = dinv[s] * dinv[d];
  const float2 v = *(const float2*)(t + (size_t)s * 64 + 2 * lane);
  float* ar = agg + (size_t)d * 64 + 2 * lane;
  atomicAdd(ar,     v.x * norm);
  atomicAdd(ar + 1, v.y * norm);
}

// h_out = [residual +] relu(agg + t*dinv^2 + bias); also stages bf16 copy.
__global__ void gcn_pointwise(const float* __restrict__ agg, const float* __restrict__ t,
                              const float* __restrict__ dinv, const float* __restrict__ bias,
                              const float* __restrict__ residual, float* __restrict__ h,
                              unsigned short* __restrict__ hbf, int total)
{
  int i = blockIdx.x * blockDim.x + threadIdx.x;
  if (i >= total) return;
  int node = i >> 6, f = i & 63;
  float di = dinv[node];
  float v = fmaxf(agg[i] + t[i] * di * di + bias[f], 0.0f);
  if (residual) v += residual[i];
  h[i] = v;
  hbf[i] = f2bf(v);
}

// ---------------------------------------------------------------------------
// GAT
// ---------------------------------------------------------------------------
// per (node, head): a_s, a_d dot products; init segment-max with self-loop e.
__global__ void gat_att(const float* __restrict__ xg, const float* __restrict__ att_src,
                        const float* __restrict__ att_dst, float* __restrict__ a_s,
                        float* __restrict__ a_d, float* __restrict__ mmax, int nTot)
{
  int i = blockIdx.x * blockDim.x + threadIdx.x;
  if (i >= nTot) return;
  int node = i >> 2, hh = i & 3;
  const float* xr = xg + (size_t)node * 256 + hh * 64;
  const float* as = att_src + hh * 64;
  const float* ad = att_dst + hh * 64;
  float sa = 0.0f, sd = 0.0f;
  for (int f = 0; f < 64; ++f) { float xv = xr[f]; sa += xv * as[f]; sd += xv * ad[f]; }
  a_s[i] = sa; a_d[i] = sd;
  mmax[i] = lrelu(sa + sd);                        // self loop always present
}

__global__ void gat_edge_max(const int* __restrict__ src, const int* __restrict__ dst,
                             const float* __restrict__ a_s, const float* __restrict__ a_d,
                             float* __restrict__ mmax, int nE4)
{
  int i = blockIdx.x * blockDim.x + threadIdx.x;
  if (i >= nE4) return;
  int e = i >> 2, hh = i & 3;
  int s = src[e], d = dst[e];
  float v = lrelu(a_s[s * 4 + hh] + a_d[d * 4 + hh]);
  atomicMax(&mmax[d * 4 + hh], v);                 // global_atomic_max_num_f32
}

// one wave per edge: 8 lanes per head, 8 features per lane (4*64 = 256)
__global__ __launch_bounds__(256) void gat_edge_acc(
    const int* __restrict__ src, const int* __restrict__ dst,
    const float* __restrict__ a_s, const float* __restrict__ a_d,
    const float* __restrict__ mmax, const float* __restrict__ xg,
    float* __restrict__ ssum, float* __restrict__ num, int nEdges)
{
  const int wave = (int)((blockIdx.x * blockDim.x + threadIdx.x) >> 5);
  const int lane = (int)(threadIdx.x & 31);
  if (wave >= nEdges) return;
  const int s = src[wave], d = dst[wave];
  const int hh = lane >> 3;
  float e  = lrelu(a_s[s * 4 + hh] + a_d[d * 4 + hh]);
  float ex = __expf(e - mmax[d * 4 + hh]);
  if ((lane & 7) == 0) atomicAdd(&ssum[d * 4 + hh], ex);
  const float* xr = xg  + (size_t)s * 256 + lane * 8;
  float*       nr = num + (size_t)d * 256 + lane * 8;
#pragma unroll
  for (int j = 0; j < 8; ++j) atomicAdd(nr + j, xr[j] * ex);
}

// hg = mean_heads((num + ex_self*xg_self) / (ssum + ex_self)) + bg; stage bf16.
__global__ void gat_finalize(const float* __restrict__ a_s, const float* __restrict__ a_d,
                             const float* __restrict__ mmax, const float* __restrict__ ssum,
                             const float* __restrict__ num, const float* __restrict__ xg,
                             const float* __restrict__ bg, float* __restrict__ hg,
                             unsigned short* __restrict__ hgbf, int total)
{
  int i = blockIdx.x * blockDim.x + threadIdx.x;
  if (i >= total) return;
  int node = i >> 6, f = i & 63;
  float acc = 0.0f;
#pragma unroll
  for (int hh = 0; hh < 4; ++hh) {
    float e   = lrelu(a_s[node * 4 + hh] + a_d[node * 4 + hh]);
    float exs = __expf(e - mmax[node * 4 + hh]);
    size_t xi = (size_t)node * 256 + hh * 64 + f;
    acc += (num[xi] + exs * xg[xi]) / (ssum[node * 4 + hh] + exs);
  }
  float v = 0.25f * acc + bg[f];
  hg[i] = v;
  hgbf[i] = f2bf(v);
}

// ---------------------------------------------------------------------------
// Pool + classifier
// ---------------------------------------------------------------------------
__global__ __launch_bounds__(64) void pool_kernel(const float* __restrict__ hg,
                                                  float* __restrict__ gmean, int nNodes)
{
  int f = threadIdx.x;
  int start = blockIdx.x * 512;
  int end = start + 512; if (end > nNodes) end = nNodes;
  float s = 0.0f;
  for (int nd = start; nd < end; ++nd) s += hg[(size_t)nd * 64 + f];
  atomicAdd(&gmean[f], s);
}

__global__ __launch_bounds__(64) void cls_kernel(const float* __restrict__ gmean,
    const float* __restrict__ Wc1, const float* __restrict__ bc1,
    const float* __restrict__ Wc2, const float* __restrict__ bc2,
    float* __restrict__ cls, float invN)
{
  __shared__ float y1[32];
  int t = threadIdx.x;
  if (t < 32) {
    float s = bc1[t];
    for (int f = 0; f < 64; ++f) s += gmean[f] * invN * Wc1[f * 32 + t];
    y1[t] = fmaxf(s, 0.0f);
  }
  __syncthreads();
  if (t < 2) {
    float s = bc2[t];
    for (int j = 0; j < 32; ++j) s += y1[j] * Wc2[j * 2 + t];
    cls[t] = s;
  }
}

// ---------------------------------------------------------------------------
static inline int ceil_div(int a, int b) { return (a + b - 1) / b; }

extern "C" void kernel_launch(void* const* d_in, const int* in_sizes, int n_in,
                              void* d_out, int out_size, void* d_ws, size_t ws_size,
                              hipStream_t stream) {
  (void)n_in; (void)out_size; (void)ws_size;

  const float* x   = (const float*)d_in[0];
  const int*   ei  = (const int*)d_in[1];        // jax default: int64 demotes to int32
  const float* W0  = (const float*)d_in[2];  const float* b0  = (const float*)d_in[3];
  const float* W1  = (const float*)d_in[4];  const float* b1  = (const float*)d_in[5];
  const float* W2  = (const float*)d_in[6];  const float* b2  = (const float*)d_in[7];
  const float* Wg  = (const float*)d_in[8];
  const float* att_src = (const float*)d_in[9];
  const float* att_dst = (const float*)d_in[10];
  const float* bg  = (const float*)d_in[11];
  const float* Wc1 = (const float*)d_in[12]; const float* bc1 = (const float*)d_in[13];
  const float* Wc2 = (const float*)d_in[14]; const float* bc2 = (const float*)d_in[15];
  const float* Wr1 = (const float*)d_in[16]; const float* br1 = (const float*)d_in[17];
  const float* Wr2 = (const float*)d_in[18]; const float* br2 = (const float*)d_in[19];

  const int N = in_sizes[0] / 256;   // 50000 (multiple of 16)
  const int E = in_sizes[1] / 2;     // 800000
  const int* src = ei;
  const int* dst = ei + E;

  // ---- workspace carve-out (256B aligned regions) ----
  char* ws = (char*)d_ws;
  size_t off = 0;
  auto alloc = [&](size_t bytes) -> char* {
    char* p = ws + off;
    off = (off + bytes + 255) & ~(size_t)255;
    return p;
  };
  float* dinv  = (float*)alloc((size_t)N * 4);
  float* t     = (float*)alloc((size_t)N * 64 * 4);
  float* agg   = (float*)alloc((size_t)N * 64 * 4);
  float* h     = (float*)alloc((size_t)N * 64 * 4);
  float* xg    = (float*)alloc((size_t)N * 256 * 4);
  float* num   = (float*)alloc((size_t)N * 256 * 4);
  float* a_s   = (float*)alloc((size_t)N * 4 * 4);
  float* a_d   = (float*)alloc((size_t)N * 4 * 4);
  float* mmax  = (float*)alloc((size_t)N * 4 * 4);
  float* ssum  = (float*)alloc((size_t)N * 4 * 4);
  float* gmean = (float*)alloc(64 * 4);
  unsigned short* xbf  = (unsigned short*)alloc((size_t)N * 256 * 2);
  unsigned short* hbf  = (unsigned short*)alloc((size_t)N * 64 * 2);
  unsigned short* hgbf = (unsigned short*)alloc((size_t)N * 64 * 2);
  unsigned short* tbf  = (unsigned short*)alloc((size_t)N * 64 * 2);
  unsigned short* W0T  = (unsigned short*)alloc((size_t)256 * 64 * 2);
  unsigned short* W1T  = (unsigned short*)alloc((size_t)64 * 64 * 2);
  unsigned short* W2T  = (unsigned short*)alloc((size_t)64 * 64 * 2);
  unsigned short* WgT  = (unsigned short*)alloc((size_t)64 * 256 * 2);
  unsigned short* Wr1T = (unsigned short*)alloc((size_t)64 * 64 * 2);
  unsigned short* Wr2T = (unsigned short*)alloc((size_t)64 * 256 * 2);

  // ---- output layout: [cls(2) | rec(N*256) | hg(N*64)] ----
  float* cls = (float*)d_out;
  float* rec = (float*)d_out + 2;
  float* hg  = (float*)d_out + 2 + (size_t)N * 256;

  // ---- degrees / normalization ----
  hipMemsetAsync(dinv, 0, (size_t)N * 4, stream);
  degree_kernel<<<ceil_div(E, 256), 256, 0, stream>>>(dst, dinv, E);
  dinv_kernel<<<ceil_div(N, 256), 256, 0, stream>>>(dinv, N);

  // ---- stage weights (bf16 transposed) and x (bf16 row-major) ----
  cvt_w_t<<<ceil_div(256 * 64, 256), 256, 0, stream>>>(W0,  W0T,  256, 64);
  cvt_w_t<<<ceil_div(64 * 64, 256), 256, 0, stream>>>(W1,  W1T,  64, 64);
  cvt_w_t<<<ceil_div(64 * 64, 256), 256, 0, stream>>>(W2,  W2T,  64, 64);
  cvt_w_t<<<ceil_div(64 * 256, 256), 256, 0, stream>>>(Wg,  WgT,  64, 256);
  cvt_w_t<<<ceil_div(64 * 64, 256), 256, 0, stream>>>(Wr1, Wr1T, 64, 64);
  cvt_w_t<<<ceil_div(64 * 256, 256), 256, 0, stream>>>(Wr2, Wr2T, 64, 256);
  cvt_bf16<<<ceil_div(N * 256, 256), 256, 0, stream>>>(x, xbf, N * 256);

  const int g64  = ceil_div((N / 16) * (64 / 16), 8);   // 16x16 tiles, kout=64
  const int g256 = ceil_div((N / 16) * (256 / 16), 8);  // kout=256

  // ---- GCN layer 0: h = relu(gcn(x, W0, b0)) ----
  gemm_wmma_bf16<256><<<g64, 256, 0, stream>>>(xbf, W0T, nullptr, t, nullptr, N, 64, 0);
  hipMemsetAsync(agg, 0, (size_t)N * 64 * 4, stream);
  gcn_edge_agg<<<ceil_div(E, 8), 256, 0, stream>>>(src, dst, t, dinv, agg, E);
  gcn_pointwise<<<ceil_div(N * 64, 256), 256, 0, stream>>>(agg, t, dinv, b0, nullptr, h, hbf, N * 64);

  // ---- GCN layer 1: h = h + relu(gcn(h, W1, b1)) ----
  gemm_wmma_bf16<64><<<g64, 256, 0, stream>>>(hbf, W1T, nullptr, t, nullptr, N, 64, 0);
  hipMemsetAsync(agg, 0, (size_t)N * 64 * 4, stream);
  gcn_edge_agg<<<ceil_div(E, 8), 256, 0, stream>>>(src, dst, t, dinv, agg, E);
  gcn_pointwise<<<ceil_div(N * 64, 256), 256, 0, stream>>>(agg, t, dinv, b1, h, h, hbf, N * 64);

  // ---- GCN layer 2: h = h + relu(gcn(h, W2, b2)) ----
  gemm_wmma_bf16<64><<<g64, 256, 0, stream>>>(hbf, W2T, nullptr, t, nullptr, N, 64, 0);
  hipMemsetAsync(agg, 0, (size_t)N * 64 * 4, stream);
  gcn_edge_agg<<<ceil_div(E, 8), 256, 0, stream>>>(src, dst, t, dinv, agg, E);
  gcn_pointwise<<<ceil_div(N * 64, 256), 256, 0, stream>>>(agg, t, dinv, b2, h, h, hbf, N * 64);

  // ---- GAT ----
  gemm_wmma_bf16<64><<<g256, 256, 0, stream>>>(hbf, WgT, nullptr, xg, nullptr, N, 256, 0);
  gat_att<<<ceil_div(N * 4, 256), 256, 0, stream>>>(xg, att_src, att_dst, a_s, a_d, mmax, N * 4);
  hipMemsetAsync(ssum, 0, (size_t)N * 4 * 4, stream);
  hipMemsetAsync(num, 0, (size_t)N * 256 * 4, stream);
  gat_edge_max<<<ceil_div(E * 4, 256), 256, 0, stream>>>(src, dst, a_s, a_d, mmax, E * 4);
  gat_edge_acc<<<ceil_div(E, 8), 256, 0, stream>>>(src, dst, a_s, a_d, mmax, xg, ssum, num, E);
  gat_finalize<<<ceil_div(N * 64, 256), 256, 0, stream>>>(a_s, a_d, mmax, ssum, num, xg, bg, hg, hgbf, N * 64);

  // ---- global mean pool + classifier ----
  hipMemsetAsync(gmean, 0, 64 * 4, stream);
  pool_kernel<<<ceil_div(N, 512), 64, 0, stream>>>(hg, gmean, N);
  cls_kernel<<<1, 64, 0, stream>>>(gmean, Wc1, bc1, Wc2, bc2, cls, 1.0f / (float)N);

  // ---- reconstructor: rec = relu(hg@Wr1+br1)@Wr2+br2 ----
  gemm_wmma_bf16<64><<<g64, 256, 0, stream>>>(hgbf, Wr1T, br1, nullptr, tbf, N, 64, 1);
  gemm_wmma_bf16<64><<<g256, 256, 0, stream>>>(tbf, Wr2T, br2, rec, nullptr, N, 256, 0);
}
